// PyramidFusion_46291157517085
// MI455X (gfx1250) — matrix operand order repaired and checked by reading, not compiled, not run on hardware
//
#include <hip/hip_runtime.h>

// ---------------------------------------------------------------------------
// PyramidFusion on MI455X (gfx1250).
// Whole graph is linear => fold into ONE 5x5 conv with composed weights:
//   W[o][i][ky][kx] = wf2*w5 + (center 3x3) wf1*w3 + (even taps) wf3*wd
//                   + (center) wf0*w1   (all matrix-composed over the middle C)
// Then implicit-GEMM 5x5 conv with v_wmma_f32_16x16x32_bf16.
// Tiling v2: 4 output rows/WG (2x halo redundancy instead of 5x),
//            wave tile M32xN64 (1 ds_load_b128 per WMMA instead of 2).
// ---------------------------------------------------------------------------

typedef __attribute__((ext_vector_type(16))) __bf16 v16bf;
typedef __attribute__((ext_vector_type(8)))  float  v8f;
typedef unsigned short u16;

#define C_IN 192
#define HW   128
#define NBATCH 8
#define ROWS  4                      // output rows per workgroup
#define TROWS (ROWS + 4)             // input tile rows (5x5, pad 2)

__device__ __forceinline__ u16 f32_bf16(float f) {
  union { float f; unsigned u; } v; v.f = f;
  unsigned r = v.u + 0x7FFFu + ((v.u >> 16) & 1u);   // round-to-nearest-even
  return (u16)(r >> 16);
}

// ---- Kernel 1: x NCHW f32  ->  NHWC bf16 (coalesced via LDS tile) ----------
__global__ void __launch_bounds__(256) k_nhwc_bf16(const float* __restrict__ x,
                                                   u16* __restrict__ xb) {
  __shared__ u16 Ls[64 * 33];
  const int row = blockIdx.x;              // b*128 + y
  const int b = row >> 7, y = row & 127;
  const int colbase = blockIdx.y * 64;
  const int chbase  = blockIdx.z * 32;
  const int tid = threadIdx.x;
#pragma unroll
  for (int i = 0; i < 8; ++i) {            // 2048 elements, 8 per thread
    int idx = tid + i * 256;
    int ch = idx >> 6, col = idx & 63;     // consecutive tid -> consecutive col
    float v = x[((b * C_IN + chbase + ch) * HW + y) * HW + colbase + col];
    Ls[col * 33 + ch] = f32_bf16(v);
  }
  __syncthreads();
#pragma unroll
  for (int i = 0; i < 2; ++i) {            // write 8B (4 channels) per task
    int idx = tid + i * 256;
    int col = idx >> 3, cg = idx & 7;
    const u16* s = &Ls[col * 33 + cg * 4];
    uint2 p;
    p.x = (unsigned)s[0] | ((unsigned)s[1] << 16);
    p.y = (unsigned)s[2] | ((unsigned)s[3] << 16);
    *(uint2*)&xb[(row * HW + colbase + col) * C_IN + chbase + cg * 4] = p;
  }
}

// ---- Kernel 2: compose effective 5x5 weights, bf16, WMMA-A-friendly layout -
// Wb[tap(25)][kc(6)][o(192)][kk(32)],  input channel i = kc*32+kk
__global__ void __launch_bounds__(256) k_weights(const float* __restrict__ w1,
                                                 const float* __restrict__ w3,
                                                 const float* __restrict__ w5,
                                                 const float* __restrict__ wd,
                                                 const float* __restrict__ wf,
                                                 u16* __restrict__ wb) {
  const int t = blockIdx.x * 256 + threadIdx.x;  // 36864 threads: (o, i)
  const int o = t / C_IN, i = t % C_IN;
  float acc[25];
#pragma unroll
  for (int k = 0; k < 25; ++k) acc[k] = 0.f;
  const float* wfo = wf + o * (4 * C_IN);
  for (int m = 0; m < C_IN; ++m) {
    const float a1 = wfo[m];
    const float a3 = wfo[C_IN + m];
    const float a5 = wfo[2 * C_IN + m];
    const float ad = wfo[3 * C_IN + m];
    const float* p5 = w5 + (m * C_IN + i) * 25;
#pragma unroll
    for (int k = 0; k < 25; ++k) acc[k] += a5 * p5[k];
    const float* p3 = w3 + (m * C_IN + i) * 9;
    const float* pd = wd + (m * C_IN + i) * 9;
#pragma unroll
    for (int r = 0; r < 3; ++r)
#pragma unroll
      for (int s = 0; s < 3; ++s) {
        acc[(r + 1) * 5 + (s + 1)] += a3 * p3[r * 3 + s];  // 3x3 centered
        acc[(2 * r) * 5 + (2 * s)] += ad * pd[r * 3 + s];  // 3x3 dilated-2
      }
    acc[12] += a1 * w1[m * C_IN + i];                       // 1x1 at center
  }
  const int kc = i >> 5, kk = i & 31;
#pragma unroll
  for (int tap = 0; tap < 25; ++tap)
    wb[(((tap * 6 + kc) * C_IN) + o) * 32 + kk] = f32_bf16(acc[tap]);
}

// ---- Kernel 3: 5x5 conv as implicit GEMM with bf16 WMMA --------------------
// WG (512 thr = 16 waves): M=64 out-channels x 4 output rows (512 pixels).
// Wave tile: M32 x N64 -> 8 accumulators; each B fragment feeds 2 WMMAs.
__global__ void __launch_bounds__(512, 1) k_conv5_wmma(const u16* __restrict__ xb,
                                                       const u16* __restrict__ wb,
                                                       float* __restrict__ out) {
  // [TROWS rows][132 cols][40 (32 ch + pad)] : col stride 80B => 16B-aligned
  // b128 ops and conflict-free half-wave B reads (bank stride 20, distinct mod 64)
  __shared__ __align__(16) u16 Xs[TROWS * 132 * 40];

  const int blk = blockIdx.x;                  // b*(128/ROWS) + yblk
  const int b = blk >> 5, y0 = (blk & 31) * ROWS;
  const int mbase = blockIdx.y * 64;
  const int tid = threadIdx.x;
  const int lane = tid & 31, wave = tid >> 5;  // 16 waves
  const int wm2 = wave & 1;                    // M half: 0/32
  const int wn8 = wave >> 1;                   // 0..7 : (row, col-half)
  const int r_out = wn8 >> 1;                  // local output row 0..3
  const int colseg = (wn8 & 1) * 64;           // 0 or 64
  const int mwave = mbase + wm2 * 32;
  const bool hi = lane >= 16;
  const int l15 = lane & 15;

  v8f acc[2][4];
#pragma unroll
  for (int mi = 0; mi < 2; ++mi)
#pragma unroll
    for (int nt = 0; nt < 4; ++nt) acc[mi][nt] = (v8f){};

  for (int kc = 0; kc < 6; ++kc) {             // K chunks of 32 input channels
    __syncthreads();
    // stage [TROWS][132][32] bf16 into LDS (16B chunks of 8 channels)
    for (int idx = tid; idx < TROWS * 132 * 4; idx += 512) {
      int chunk = idx & 3;
      int pos = idx >> 2;
      int c = pos % 132, r = pos / 132;
      int yy = y0 + r - 2, xx = c - 2;
      uint4 v = {0u, 0u, 0u, 0u};
      if (yy >= 0 && yy < HW && xx >= 0 && xx < HW)
        v = *(const uint4*)&xb[((b * HW + yy) * HW + xx) * C_IN + kc * 32 + chunk * 8];
      *(uint4*)&Xs[(r * 132 + c) * 40 + chunk * 8] = v;
    }
    __syncthreads();

    for (int ky = 0; ky < 5; ++ky) {
#pragma unroll
      for (int kx = 0; kx < 5; ++kx) {
        const int tap = ky * 5 + kx;
        // A fragments: lane<16 holds K0..15, lane>=16 K16..31
        const u16* arow0 = wb + (((tap * 6 + kc) * C_IN) + mwave + l15) * 32;
        union { uint4 q[2]; v16bf v; } A0, A1;
        A0.q[0] = *(const uint4*)(arow0 + (hi ? 16 : 0));
        A0.q[1] = *(const uint4*)(arow0 + (hi ? 24 : 8));
        A1.q[0] = *(const uint4*)(arow0 + 512 + (hi ? 16 : 0));    // m + 16
        A1.q[1] = *(const uint4*)(arow0 + 512 + (hi ? 24 : 8));
#pragma unroll
        for (int nt = 0; nt < 4; ++nt) {
          // B fragment: N=lane&15, lanes<16 K0..15 / lanes>=16 K16..31,
          // 32B contiguous in LDS (channels-contiguous layout)
          const int col = colseg + nt * 16 + l15;
          const uint4* bp =
              (const uint4*)&Xs[((r_out + ky) * 132 + col + kx) * 40 + (hi ? 16 : 0)];
          union { uint4 q[2]; v16bf v; } B;
          B.q[0] = bp[0];
          B.q[1] = bp[1];
          acc[0][nt] = __builtin_amdgcn_wmma_f32_16x16x32_bf16(
              false, A0.v, false, B.v, (short)0, acc[0][nt], false, false);
          acc[1][nt] = __builtin_amdgcn_wmma_f32_16x16x32_bf16(
              false, A1.v, false, B.v, (short)0, acc[1][nt], false, false);
        }
      }
    }
  }

  // epilogue: C/D layout -> NCHW f32. VGPR j: M = j + (lane<16?0:8), N = lane&15
  const int ycur = y0 + r_out;
#pragma unroll
  for (int mi = 0; mi < 2; ++mi) {
    const int obase = mwave + mi * 16 + (hi ? 8 : 0);
#pragma unroll
    for (int nt = 0; nt < 4; ++nt) {
      const int col = colseg + nt * 16 + l15;
#pragma unroll
      for (int j = 0; j < 8; ++j) {
        const int o = obase + j;
        out[((b * C_IN + o) * HW + ycur) * HW + col] = acc[mi][nt][j];
      }
    }
  }
}

// ---------------------------------------------------------------------------
extern "C" void kernel_launch(void* const* d_in, const int* in_sizes, int n_in,
                              void* d_out, int out_size, void* d_ws, size_t ws_size,
                              hipStream_t stream) {
  const float* x  = (const float*)d_in[0];
  const float* w1 = (const float*)d_in[1];
  const float* w3 = (const float*)d_in[2];
  const float* w5 = (const float*)d_in[3];
  const float* wd = (const float*)d_in[4];
  const float* wf = (const float*)d_in[5];
  float* out = (float*)d_out;

  // workspace layout: NHWC bf16 x (50,331,648 B) then composed bf16 weights
  u16* xb = (u16*)d_ws;
  u16* wb = (u16*)((char*)d_ws + (size_t)NBATCH * HW * HW * C_IN * sizeof(u16));

  // 1) x -> NHWC bf16
  k_nhwc_bf16<<<dim3(NBATCH * HW, 2, 6), 256, 0, stream>>>(x, xb);
  // 2) compose 5x5 effective weights (192*192 threads)
  k_weights<<<(C_IN * C_IN) / 256, 256, 0, stream>>>(w1, w3, w5, wd, wf, wb);
  // 3) single fused 5x5 conv via WMMA (4 rows per WG, 512 threads)
  k_conv5_wmma<<<dim3(NBATCH * HW / ROWS, 3), 512, 0, stream>>>(xb, wb, out);
}
